// Transformer_21242908246230
// MI455X (gfx1250) — matrix operand order repaired and checked by reading, not compiled, not run on hardware
//
#include <hip/hip_runtime.h>

typedef __attribute__((ext_vector_type(16))) _Float16 v16h;
typedef __attribute__((ext_vector_type(8)))  _Float16 v8h;
typedef __attribute__((ext_vector_type(8)))  float    v8f;
typedef __attribute__((ext_vector_type(4)))  float    v4f;
typedef __attribute__((ext_vector_type(4)))  int      v4i;

#define S_TOTAL 16384
#define Q_TOTAL 4096
#define D_IN    128
#define DK      64
#define DV      64
#define NWAVE   8
#define S_CHUNK (S_TOTAL / NWAVE)   // 2048 per wave

// logits are kept in log2 domain: scale = (1/sqrt(64)) * log2(e)
#define SCALE_LOG2 0.18033688011112042f
#define NEG_BIG    (-3.0e38f)

// ---------------- helpers ----------------

__device__ __forceinline__ v8f wmma_f16(v16h a, v16h b, v8f c) {
  // D = A(16x32 f16) x B(32x16 f16) + C(16x16 f32)
  return __builtin_amdgcn_wmma_f32_16x16x32_f16(false, a, false, b, (short)0, c,
                                                false, false);
}

// A-operand loader: matrix rows are 64 f16, lane m = lane&15 is the M row,
// K-chunk c in {0,32}. Contiguous layout: elems 0..15 = K = c + h*16 + {0..15}.
__device__ __forceinline__ v16h ldA(const _Float16* __restrict__ row, int c, int h) {
  v8h lo = *(const v8h*)(row + c + h * 16);
  v8h hi = *(const v8h*)(row + c + h * 16 + 8);
  return __builtin_shufflevector(lo, hi, 0, 1, 2, 3, 4, 5, 6, 7,
                                         8, 9, 10, 11, 12, 13, 14, 15);
}

// ---------------- kernel 1: linear projections to f16 ----------------
// out[row][j] = sum_k x[row][k] * W[k][j] + b[j], 4 rows per 256-thread block.
__global__ void proj_kernel(const float* __restrict__ x,
                            const float* __restrict__ W,
                            const float* __restrict__ b,
                            _Float16* __restrict__ out) {
  __shared__ float xs[4 * D_IN];
  const int t = threadIdx.x;
  const int row0 = blockIdx.x * 4;
  xs[t]       = x[(size_t)row0 * D_IN + t];
  xs[t + 256] = x[(size_t)row0 * D_IN + t + 256];
  __syncthreads();
  const int r = t >> 6, j = t & 63;
  const float* xr = &xs[r * D_IN];
  float acc = 0.f;
#pragma unroll 16
  for (int k = 0; k < D_IN; ++k) acc = fmaf(xr[k], W[k * DK + j], acc);
  acc += b[j];
  out[(size_t)(row0 + r) * DK + j] = (_Float16)acc;
}

// ---------------- kernel 2: V -> Vt (f16, [DV][S]) ----------------
__global__ void transpose_v_kernel(const float* __restrict__ y,
                                   _Float16* __restrict__ vt) {
  __shared__ _Float16 tile[64][65];
  const int t = threadIdx.x;
  const int s0 = blockIdx.x * 64;
#pragma unroll
  for (int i = 0; i < 16; ++i) {
    int idx = t + 256 * i;           // 64 s x 64 dv
    int sl = idx >> 6, dv = idx & 63;
    tile[sl][dv] = (_Float16)y[(size_t)(s0 + sl) * DV + dv];
  }
  __syncthreads();
#pragma unroll
  for (int i = 0; i < 16; ++i) {
    int idx = t + 256 * i;
    int dv = idx >> 6, sl = idx & 63;
    vt[(size_t)dv * S_TOTAL + s0 + sl] = tile[sl][dv];
  }
}

// ---------------- kernel 3: flash attention ----------------
// Block: 256 threads = 8 waves. Block handles 16 queries; wave w handles
// S range [w*2048, (w+1)*2048) with online softmax (log2 domain), partials
// merged in LDS.
__global__ __launch_bounds__(256)
void attn_kernel(const float* __restrict__ soft_mask,
                 const int* __restrict__ mask,
                 const _Float16* __restrict__ keys,
                 const _Float16* __restrict__ queries,
                 const _Float16* __restrict__ vt,
                 float* __restrict__ out) {
  __shared__ float smax[NWAVE * 16];
  __shared__ float ssum[NWAVE * 16];
  __shared__ float sacc[DV * 16];
  __shared__ float sginv[16];

  const int t = threadIdx.x;
  const int lane = t & 31, wav = t >> 5;
  const int n = lane & 15;     // q column for B/C operands, M row for A
  const int h = lane >> 4;     // lane half
  const int q0 = blockIdx.x * 16;

#pragma unroll
  for (int i = 0; i < 4; ++i) sacc[t + 256 * i] = 0.f;

  // Qt B-operand (32x16): lane n = query col; elems 0..15 = K = c + h*16 + {0..15}
  const _Float16* qrow = queries + (size_t)(q0 + n) * DK;
  const v16h bq0 = *(const v16h*)(qrow + h * 16);
  const v16h bq1 = *(const v16h*)(qrow + 32 + h * 16);

  const v8f zero8 = {0.f, 0.f, 0.f, 0.f, 0.f, 0.f, 0.f, 0.f};
  v8f acc[4];  // out^T tiles: dv = 16*d + r + 8*h, q = n
#pragma unroll
  for (int d = 0; d < 4; ++d) acc[d] = zero8;

  float runmax = NEG_BIG, runsum = 0.f;   // log2-domain running max

  const float* smrow = soft_mask + (size_t)(q0 + n) * S_TOTAL;
  const int*   mkrow = mask      + (size_t)(q0 + n) * S_TOTAL;

  const int sBeg = wav * S_CHUNK;
  const int sEnd = sBeg + S_CHUNK;

  for (int sT = sBeg; sT < sEnd; sT += 32) {
    // ---- L^T tiles: c0 = K[sT..sT+15] x Q^T, c1 = K[sT+16..sT+31] x Q^T ----
    v8f c0 = zero8, c1 = zero8;
    {
      const _Float16* k0 = keys + (size_t)(sT + n) * DK;
      const _Float16* k1 = keys + (size_t)(sT + 16 + n) * DK;
      c0 = wmma_f16(ldA(k0, 0, h),  bq0, c0);
      c0 = wmma_f16(ldA(k0, 32, h), bq1, c0);
      c1 = wmma_f16(ldA(k1, 0, h),  bq0, c1);
      c1 = wmma_f16(ldA(k1, 32, h), bq1, c1);
    }

    // ---- stream soft_mask / mask (non-temporal; 8 consecutive s per lane) ----
    v4f sm0a = __builtin_nontemporal_load((const v4f*)(smrow + sT + 8 * h));
    v4f sm0b = __builtin_nontemporal_load((const v4f*)(smrow + sT + 8 * h + 4));
    v4f sm1a = __builtin_nontemporal_load((const v4f*)(smrow + sT + 16 + 8 * h));
    v4f sm1b = __builtin_nontemporal_load((const v4f*)(smrow + sT + 16 + 8 * h + 4));
    v4i mk0a = __builtin_nontemporal_load((const v4i*)(mkrow + sT + 8 * h));
    v4i mk0b = __builtin_nontemporal_load((const v4i*)(mkrow + sT + 8 * h + 4));
    v4i mk1a = __builtin_nontemporal_load((const v4i*)(mkrow + sT + 16 + 8 * h));
    v4i mk1b = __builtin_nontemporal_load((const v4i*)(mkrow + sT + 16 + 8 * h + 4));

    // ---- masked log2-logits + tile max ----
    float lg0[8], lg1[8];
    float mloc = NEG_BIG;
#pragma unroll
    for (int r = 0; r < 8; ++r) {
      int m0 = (r < 4) ? mk0a[r] : mk0b[r - 4];
      int m1 = (r < 4) ? mk1a[r] : mk1b[r - 4];
      lg0[r] = m0 ? (c0[r] * SCALE_LOG2) : NEG_BIG;
      lg1[r] = m1 ? (c1[r] * SCALE_LOG2) : NEG_BIG;
      mloc = fmaxf(mloc, fmaxf(lg0[r], lg1[r]));
    }
    mloc = fmaxf(mloc, __shfl_xor(mloc, 16, 32));  // combine halves (same q)
    const float nm = fmaxf(runmax, mloc);
    const float fac = __builtin_amdgcn_exp2f(runmax - nm);
    runmax = nm;

    // ---- e = 2^(lg - nm) * soft_mask * mask ; row partial sums ----
    float e0[8], e1[8];
    float lsum = 0.f;
#pragma unroll
    for (int r = 0; r < 8; ++r) {
      float s0v = (r < 4) ? sm0a[r] : sm0b[r - 4];
      float s1v = (r < 4) ? sm1a[r] : sm1b[r - 4];
      int m0 = (r < 4) ? mk0a[r] : mk0b[r - 4];
      int m1 = (r < 4) ? mk1a[r] : mk1b[r - 4];
      e0[r] = m0 ? (__builtin_amdgcn_exp2f(lg0[r] - nm) * s0v) : 0.f;
      e1[r] = m1 ? (__builtin_amdgcn_exp2f(lg1[r] - nm) * s1v) : 0.f;
      lsum += e0[r] + e1[r];
    }
    lsum += __shfl_xor(lsum, 16, 32);
    runsum = runsum * fac + lsum;
#pragma unroll
    for (int d = 0; d < 4; ++d)
#pragma unroll
      for (int r = 0; r < 8; ++r) acc[d][r] *= fac;

    // ---- build P^T B-operand (32x16) in registers: pack + cross-half swap ----
    union { unsigned u[8]; v16h v; } bp;
#pragma unroll
    for (int v = 0; v < 4; ++v) {
      auto pa = __builtin_amdgcn_cvt_pkrtz(e0[2 * v], e0[2 * v + 1]);
      auto pb = __builtin_amdgcn_cvt_pkrtz(e1[2 * v], e1[2 * v + 1]);
      unsigned ua = __builtin_bit_cast(unsigned, pa);
      unsigned ub = __builtin_bit_cast(unsigned, pb);
      unsigned xa = (unsigned)__shfl_xor((int)ua, 16, 32);
      unsigned xb = (unsigned)__shfl_xor((int)ub, 16, 32);
      bp.u[v]     = h ? xb : ua;  // K = h*16 + 2v   -> tile[h] rows 2v,2v+1
      bp.u[v + 4] = h ? ub : xa;  // K = h*16 + 8+2v -> tile[h] rows 8+2v,..
    }

    // ---- out^T += V^T[16d.., sT..sT+31] x P^T ----
#pragma unroll
    for (int d = 0; d < 4; ++d) {
      const _Float16* vrow = vt + (size_t)(16 * d + n) * S_TOTAL + sT;
      v8h lo = *(const v8h*)(vrow + 16 * h);
      v8h hi = *(const v8h*)(vrow + 16 * h + 8);
      v16h av = __builtin_shufflevector(lo, hi, 0, 1, 2, 3, 4, 5, 6, 7,
                                                8, 9, 10, 11, 12, 13, 14, 15);
      acc[d] = wmma_f16(av, bp.v, acc[d]);
    }
  }

  // ---- merge 8 waves' partials (log2 domain) ----
  if (h == 0) {
    smax[wav * 16 + n] = runmax;
    ssum[wav * 16 + n] = runsum;
  }
  __syncthreads();

  float gm = NEG_BIG;
#pragma unroll
  for (int w = 0; w < NWAVE; ++w) gm = fmaxf(gm, smax[w * 16 + n]);
  const float f = __builtin_amdgcn_exp2f(runmax - gm);
#pragma unroll
  for (int d = 0; d < 4; ++d)
#pragma unroll
    for (int r = 0; r < 8; ++r)
      atomicAdd(&sacc[(16 * d + r + 8 * h) * 16 + n], acc[d][r] * f);

  if (t < 16) {
    float gm2 = NEG_BIG;
#pragma unroll
    for (int w = 0; w < NWAVE; ++w) gm2 = fmaxf(gm2, smax[w * 16 + t]);
    float gs = 0.f;
#pragma unroll
    for (int w = 0; w < NWAVE; ++w)
      gs += ssum[w * 16 + t] * __builtin_amdgcn_exp2f(smax[w * 16 + t] - gm2);
    sginv[t] = 1.0f / gs;
  }
  __syncthreads();

  // out[q0+q][0][dv] — coalesced store
#pragma unroll
  for (int i = 0; i < 4; ++i) {
    int idx = t + 256 * i;            // 16 q x 64 dv
    int q = idx >> 6, dv = idx & 63;
    out[(size_t)(q0 + q) * DV + dv] = sacc[dv * 16 + q] * sginv[q];
  }
}

// ---------------- host ----------------
extern "C" void kernel_launch(void* const* d_in, const int* in_sizes, int n_in,
                              void* d_out, int out_size, void* d_ws, size_t ws_size,
                              hipStream_t stream) {
  const float* search_x  = (const float*)d_in[0];
  const float* search_y  = (const float*)d_in[1];
  const float* query_x   = (const float*)d_in[2];
  const float* soft_mask = (const float*)d_in[3];
  const int*   mask      = (const int*)d_in[4];
  const float* W_key     = (const float*)d_in[5];
  const float* b_key     = (const float*)d_in[6];
  const float* W_query   = (const float*)d_in[7];
  const float* b_query   = (const float*)d_in[8];
  float* out = (float*)d_out;

  _Float16* keys    = (_Float16*)d_ws;                       // [S][64]
  _Float16* queries = keys + (size_t)S_TOTAL * DK;           // [Q][64]
  _Float16* vt      = queries + (size_t)Q_TOTAL * DK;        // [64][S]

  proj_kernel<<<S_TOTAL / 4, 256, 0, stream>>>(search_x, W_key, b_key, keys);
  proj_kernel<<<Q_TOTAL / 4, 256, 0, stream>>>(query_x, W_query, b_query, queries);
  transpose_v_kernel<<<S_TOTAL / 64, 256, 0, stream>>>(search_y, vt);
  attn_kernel<<<Q_TOTAL / 16, 256, 0, stream>>>(soft_mask, mask, keys, queries, vt, out);
}
